// RNN_63720134803696
// MI455X (gfx1250) — matrix operand order repaired and checked by reading, not compiled, not run on hardware
//
#include <hip/hip_runtime.h>
#include <hip/hip_bf16.h>
#include <math.h>

// Problem constants (match reference)
#define L_ 512
#define B_ 128
#define D_ 512
#define H_ 512

typedef __attribute__((ext_vector_type(16))) _Float16 v16h;
typedef __attribute__((ext_vector_type(8)))  _Float16 v8h;
typedef __attribute__((ext_vector_type(8)))  float    v8f;
typedef __attribute__((ext_vector_type(4)))  float    v4f;

typedef unsigned int u32x4 __attribute__((ext_vector_type(4)));
typedef int          i32x4 __attribute__((ext_vector_type(4)));
typedef int          i32x8 __attribute__((ext_vector_type(8)));

// ---- workspace layout (bytes) ----
#define OFF_CTR   0         // int barrier counter
#define OFF_BIAS  1024      // H_ floats  (Wx_b + Wh_b)
#define OFF_WX16  4096      // H_*D_ f16  (512 KB)
#define OFF_WH16  528384    // H_*H_ f16  (512 KB)
#define OFF_H16   1052672   // B_*H_ f16  (128 KB)  -- current hidden state, f16

#define NWG_SCAN 16         // workgroups in the scan kernel (H_/32 slices)

__device__ __forceinline__ v16h combine16(v8h lo, v8h hi) {
  v16h r;
#pragma unroll
  for (int i = 0; i < 8; ++i) { r[i] = lo[i]; r[i + 8] = hi[i]; }
  return r;
}

// -------------------------------------------------------------------------
// Kernel 0: convert weights + h0 to f16, fold biases, reset grid barrier.
// -------------------------------------------------------------------------
__global__ __launch_bounds__(256) void k_setup(
    const float* __restrict__ Wx_w, const float* __restrict__ Wx_b,
    const float* __restrict__ Wh_w, const float* __restrict__ Wh_b,
    const float* __restrict__ h0, unsigned char* __restrict__ ws) {
  int*      ctr  = (int*)(ws + OFF_CTR);
  float*    bias = (float*)(ws + OFF_BIAS);
  _Float16* wx16 = (_Float16*)(ws + OFF_WX16);
  _Float16* wh16 = (_Float16*)(ws + OFF_WH16);
  _Float16* h16  = (_Float16*)(ws + OFF_H16);

  int tid = blockIdx.x * blockDim.x + threadIdx.x;
  int nth = gridDim.x * blockDim.x;
  if (tid == 0) *ctr = 0;
  for (int i = tid; i < H_; i += nth)      bias[i] = Wx_b[i] + Wh_b[i];
  for (int i = tid; i < H_ * D_; i += nth) wx16[i] = (_Float16)Wx_w[i];
  for (int i = tid; i < H_ * H_; i += nth) wh16[i] = (_Float16)Wh_w[i];
  for (int i = tid; i < B_ * H_; i += nth) h16[i]  = (_Float16)h0[i];
}

// -------------------------------------------------------------------------
// Kernel 1: xproj = x @ Wx^T + bias, written into d_out's hAll region.
// Each wave: 32 rows (of M = L*B) x 64 cols (of H). WMMA f16 -> f32.
// A (x) converted fp32->f16 on the fly; B (Wx16) loaded from ws (L2-resident).
// -------------------------------------------------------------------------
__global__ __launch_bounds__(256) void k_xproj(
    const float* __restrict__ x, const unsigned char* __restrict__ ws,
    float* __restrict__ out) {
  const _Float16* wx16 = (const _Float16*)(ws + OFF_WX16);
  const float*    bias = (const float*)(ws + OFF_BIAS);

  const int wid  = (blockIdx.x * blockDim.x + threadIdx.x) >> 5;  // 0..16383
  const int lane = threadIdx.x & 31;
  const int mi   = wid >> 3;      // 0..2047 : 32-row block
  const int ni   = wid & 7;       // 0..7    : 64-col block
  const int row0 = mi * 32;
  const int col0 = ni * 64;

  const int r     = lane & 15;
  const int hi    = lane >> 4;
  const int koffA = hi * 8;       // A: K-halves offset
  const int koffB = hi * 8;       // B: K-halves offset

  v8f acc[2][4];
#pragma unroll
  for (int mt = 0; mt < 2; ++mt)
#pragma unroll
    for (int nt = 0; nt < 4; ++nt) acc[mt][nt] = (v8f)(0.0f);

  for (int kb = 0; kb < D_; kb += 32) {
    // --- A: two 16x32 f16 tiles, converted from fp32 x on the fly ---
    v16h a[2];
#pragma unroll
    for (int mt = 0; mt < 2; ++mt) {
      const float* ap = x + (size_t)(row0 + mt * 16 + r) * D_ + kb;
      v4f f0 = *(const v4f*)(ap + koffA);
      v4f f1 = *(const v4f*)(ap + koffA + 4);
      v4f f2 = *(const v4f*)(ap + 16 + koffA);
      v4f f3 = *(const v4f*)(ap + 16 + koffA + 4);
      v16h av;
#pragma unroll
      for (int i = 0; i < 4; ++i) {
        av[i]      = (_Float16)f0[i];
        av[4 + i]  = (_Float16)f1[i];
        av[8 + i]  = (_Float16)f2[i];
        av[12 + i] = (_Float16)f3[i];
      }
      a[mt] = av;
    }
    // --- B: four 32x16 tiles from Wx16 [N,K] row-major ---
#pragma unroll
    for (int nt = 0; nt < 4; ++nt) {
      const _Float16* bp = wx16 + (size_t)(col0 + nt * 16 + r) * D_ + kb + koffB;
      v16h bv = combine16(*(const v8h*)bp, *(const v8h*)(bp + 16));
      acc[0][nt] = __builtin_amdgcn_wmma_f32_16x16x32_f16(
          false, a[0], false, bv, (short)0, acc[0][nt], false, false);
      acc[1][nt] = __builtin_amdgcn_wmma_f32_16x16x32_f16(
          false, a[1], false, bv, (short)0, acc[1][nt], false, false);
    }
  }

  // --- epilogue: + bias, store fp32 into hAll region ---
#pragma unroll
  for (int nt = 0; nt < 4; ++nt) {
    const int col = col0 + nt * 16 + r;   // C: n = lane&15
    const float bv = bias[col];
#pragma unroll
    for (int mt = 0; mt < 2; ++mt) {
#pragma unroll
      for (int v = 0; v < 8; ++v) {
        const int m = v + 8 * hi;         // C: m = vgpr + 8*(lane>>4)
        out[(size_t)(row0 + mt * 16 + m) * H_ + col] = acc[mt][nt][v] + bv;
      }
    }
  }
}

// -------------------------------------------------------------------------
// Kernel 2: sequential scan. 16 WGs, each owns 32 columns of H.
// Wh slice (32x512 f16 = 32 KB) lives in LDS (padded, conflict-free),
// preloaded by the Tensor Data Mover (2-D descriptor with hardware LDS
// padding: +16 B after every 1024 B row -> 520-half row stride).
// Per step: each of 8 waves computes a 16(batch) x 32(col) tile,
// h_new = tanh(acc + xproj) overwrites hAll[t] in place; f16 copy to h16.
// Grid-wide sync: monotonic atomic counting barrier.
// -------------------------------------------------------------------------
__global__ __launch_bounds__(256) void k_scan(
    unsigned char* __restrict__ ws, float* __restrict__ out) {
  const _Float16* wh16 = (const _Float16*)(ws + OFF_WH16);
  _Float16*       h16  = (_Float16*)(ws + OFF_H16);
  int*            ctr  = (int*)(ws + OFF_CTR);

  // 32 rows (cols of the output) x 512 K, padded to 520 halves:
  // row stride = 260 dwords, 260 % 64 = 4 banks -> 16 lanes hit 16 disjoint
  // 4-bank groups for b128 reads. 33,280 B of LDS.
  __shared__ _Float16 ldsW[32][520];

  const int wg    = blockIdx.x;            // 0..15
  const int col0  = wg * 32;
  const int wave  = threadIdx.x >> 5;      // 0..7
  const int lane  = threadIdx.x & 31;
  const int r     = lane & 15;
  const int hi    = lane >> 4;
  const int koffA = hi * 8;
  const int koffB = hi * 8;
  const int rb    = wave * 16;             // batch-row base for this wave

  // ---- TDM: DMA the Wh slice (32 rows x 512 halves) into LDS ----------
#if __has_builtin(__builtin_amdgcn_tensor_load_to_lds)
  if (threadIdx.x < 32) {  // one wave issues the descriptor (EXEC ignored)
    const unsigned lds_addr =
        (unsigned)(unsigned long long)(uintptr_t)&ldsW[0][0];
    const unsigned long long ga =
        (unsigned long long)(uintptr_t)(wh16 + (size_t)col0 * H_);
    // D# group 0: count=1 | lds_addr | 57-bit global addr | type=2
    u32x4 g0 = {1u, lds_addr, (unsigned)(ga & 0xFFFFFFFFu),
                (unsigned)((ga >> 32) & 0x01FFFFFFu) | (2u << 30)};
    // D# group 1: data_size=2B; pad_enable, pad_interval=7 (1024B),
    // pad_amount=3 (16B); tensor_dim0=512, tensor_dim1=32;
    // tile_dim0=512, tile_dim1=32; tensor_dim0_stride=512.
    i32x8 g1 = {0x07D10000, 512 << 16, 32 << 16, 512 << 16, 32, 512, 0, 0};
    i32x4 gz = {0, 0, 0, 0};
#if __clang_major__ >= 23
    i32x8 gz8 = {0, 0, 0, 0, 0, 0, 0, 0};
    __builtin_amdgcn_tensor_load_to_lds(g0, g1, gz, gz, gz8, 0);
#else
    __builtin_amdgcn_tensor_load_to_lds(g0, g1, gz, gz, 0);
#endif
  }
#if __has_builtin(__builtin_amdgcn_s_wait_tensorcnt)
  __builtin_amdgcn_s_wait_tensorcnt((short)0);
#else
  asm volatile("s_wait_tensorcnt 0x0" ::: "memory");
#endif
#endif  // tensor_load_to_lds

  // Correctness net: cooperative b128 copy of the same bytes (L2-hot after
  // the TDM prefetch; also covers toolchains without the TDM builtin).
  for (int c = threadIdx.x; c < 2048; c += 256) {
    const int jl = c >> 6;                 // local col 0..31
    const int kc = (c & 63) * 8;           // K offset in halves
    *(v8h*)&ldsW[jl][kc] = *(const v8h*)(wh16 + (size_t)(col0 + jl) * H_ + kc);
  }
  __syncthreads();

  float* hlast = out + (size_t)L_ * B_ * H_;

  for (int t = 0; t < L_; ++t) {
    v8f acc0 = (v8f)(0.0f), acc1 = (v8f)(0.0f);

    for (int kb = 0; kb < H_; kb += 32) {
      // A: 16x32 f16 tile of the current hidden state
      const _Float16* ap = h16 + (size_t)(rb + r) * H_ + kb;
      v16h av = combine16(*(const v8h*)(ap + koffA),
                          *(const v8h*)(ap + 16 + koffA));
      // B: two 32x16 tiles from LDS
      const _Float16* bp0 = &ldsW[0 + r][kb + koffB];
      const _Float16* bp1 = &ldsW[16 + r][kb + koffB];
      v16h bv0 = combine16(*(const v8h*)bp0, *(const v8h*)(bp0 + 16));
      v16h bv1 = combine16(*(const v8h*)bp1, *(const v8h*)(bp1 + 16));
      acc0 = __builtin_amdgcn_wmma_f32_16x16x32_f16(
          false, av, false, bv0, (short)0, acc0, false, false);
      acc1 = __builtin_amdgcn_wmma_f32_16x16x32_f16(
          false, av, false, bv1, (short)0, acc1, false, false);
    }

    // epilogue: h_new = tanh(acc + xproj); overwrite hAll[t]; f16 copy.
    float* hall_t = out + (size_t)t * (B_ * H_);
#pragma unroll
    for (int ct = 0; ct < 2; ++ct) {
      const int col = col0 + ct * 16 + r;
      const v8f a = ct ? acc1 : acc0;
#pragma unroll
      for (int v = 0; v < 8; ++v) {
        const int m   = v + 8 * hi;
        const int row = rb + m;
        const size_t idx = (size_t)row * H_ + col;
        const float val = tanhf(a[v] + hall_t[idx]);
        hall_t[idx] = val;
        h16[idx]    = (_Float16)val;
        if (t == L_ - 1) hlast[idx] = val;
      }
    }

    // ---- grid-wide barrier (monotonic count: deterministic per launch) ----
    __threadfence();
    __syncthreads();
    if (threadIdx.x == 0) {
      __hip_atomic_fetch_add(ctr, 1, __ATOMIC_ACQ_REL, __HIP_MEMORY_SCOPE_AGENT);
      const int target = NWG_SCAN * (t + 1);
      while (__hip_atomic_load(ctr, __ATOMIC_ACQUIRE, __HIP_MEMORY_SCOPE_AGENT) <
             target) {
        __builtin_amdgcn_s_sleep(2);
      }
    }
    __syncthreads();
    __threadfence();  // pull fresh h16 from other WGPs on next iteration
  }
}

// -------------------------------------------------------------------------
extern "C" void kernel_launch(void* const* d_in, const int* in_sizes, int n_in,
                              void* d_out, int out_size, void* d_ws,
                              size_t ws_size, hipStream_t stream) {
  const float* x    = (const float*)d_in[0];  // [L,B,D]
  const float* h0   = (const float*)d_in[1];  // [B,H]
  const float* Wx_w = (const float*)d_in[2];  // [H,D]
  const float* Wx_b = (const float*)d_in[3];  // [H]
  const float* Wh_w = (const float*)d_in[4];  // [H,H]
  const float* Wh_b = (const float*)d_in[5];  // [H]
  float* out = (float*)d_out;                 // hAll [L,B,H] ++ h_last [B,H]
  unsigned char* ws = (unsigned char*)d_ws;

  hipLaunchKernelGGL(k_setup, dim3(256), dim3(256), 0, stream,
                     Wx_w, Wx_b, Wh_w, Wh_b, h0, ws);
  // 16384 waves = 2048 row-blocks x 8 col-blocks, 8 waves per 256-thread WG
  hipLaunchKernelGGL(k_xproj, dim3(2048), dim3(256), 0, stream, x, ws, out);
  hipLaunchKernelGGL(k_scan, dim3(NWG_SCAN), dim3(256), 0, stream, ws, out);
}
